// ViLViewLayer_35802847380325
// MI455X (gfx1250) — matrix operand order repaired and checked by reading, not compiled, not run on hardware
//
#include <hip/hip_runtime.h>
#include <hip/hip_bf16.h>
#include <math.h>

typedef __attribute__((ext_vector_type(16))) _Float16 v16h;
typedef __attribute__((ext_vector_type(8)))  _Float16 v8h;
typedef __attribute__((ext_vector_type(8)))  float    v8f;

namespace vil {

constexpr int Bv = 2, NFr = 8, DIMc = 128, Hh = 40, Wd = 40;
constexpr int HID = 256, INNER = 512, NH = 128;
constexpr int NPIX = Bv * Hh * Wd;       // 3200
constexpr int M    = NPIX * NFr;         // 25600 tokens

// register blocking: each wave computes (MT*16) x (NT*16) of output
constexpr int MT = 2, NT = 4;            // 32 x 64 per wave, 8 WMMAs per K-step

__device__ __forceinline__ float geluf(float x) {
  return 0.5f * x * (1.f + erff(x * 0.70710678118654752f));
}
__device__ __forceinline__ float siluf(float x) {
  return x / (1.f + expf(-x));
}

// ---------------------------------------------------------------------------
// Blocked WMMA macro-tile: acc[i][j] += A(32xK,f16) * W(NxK,f16)^T  (16x16 tiles)
// A row-major (M,K), W row-major (N,K). Fragments use contiguous k = 16*half + s for both A and B.
// ---------------------------------------------------------------------------
__device__ __forceinline__ void gemm_block(const _Float16* __restrict__ A,
                                           const _Float16* __restrict__ W,
                                           int m0, int n0, int K,
                                           v8f (&acc)[MT][NT]) {
  const int lane = threadIdx.x & 31;
  const int hsel = lane >> 4;      // 0: lanes 0-15, 1: lanes 16-31
  const int r    = lane & 15;
  const _Float16* arow[MT];
  const _Float16* brow[NT];
#pragma unroll
  for (int i = 0; i < MT; i++)
    arow[i] = A + (size_t)(m0 + i * 16 + r) * K + hsel * 16;  // A: K {0-15}/{16-31}
#pragma unroll
  for (int j = 0; j < NT; j++)
    brow[j] = W + (size_t)(n0 + j * 16 + r) * K + hsel * 16;  // B: K {0-15}/{16-31}

  v8f zero = {};
#pragma unroll
  for (int i = 0; i < MT; i++)
#pragma unroll
    for (int j = 0; j < NT; j++) acc[i][j] = zero;

  for (int k = 0; k < K; k += 32) {
    v16h a[MT], b[NT];
#pragma unroll
    for (int i = 0; i < MT; i++) {
      v8h alo = *(const v8h*)(arow[i] + k);
      v8h ahi = *(const v8h*)(arow[i] + k + 8);
      a[i] = __builtin_shufflevector(alo, ahi,
                                     0, 1, 2, 3, 4, 5, 6, 7,
                                     8, 9, 10, 11, 12, 13, 14, 15);
    }
#pragma unroll
    for (int j = 0; j < NT; j++) b[j] = *(const v16h*)(brow[j] + k);
#pragma unroll
    for (int i = 0; i < MT; i++)
#pragma unroll
      for (int j = 0; j < NT; j++)
        acc[i][j] = __builtin_amdgcn_wmma_f32_16x16x32_f16(false, a[i], false, b[j],
                                                           (short)0, acc[i][j],
                                                           false, false);
  }
}

// ---------------------------------------------------------------------------
// Small elementwise helpers
// ---------------------------------------------------------------------------
__global__ void k_cvt16(const float* __restrict__ s, _Float16* __restrict__ d, int n) {
  int i = blockIdx.x * blockDim.x + threadIdx.x;
  if (i < n) d[i] = (_Float16)s[i];
}
__global__ void k_copy_f32(const float* __restrict__ s, float* __restrict__ d, int n) {
  int i = blockIdx.x * blockDim.x + threadIdx.x;
  if (i < n) d[i] = s[i];
}

// x(B,NF,C,H,W) -> x_flat(M=pix*S, C) as f16
__global__ void k_pack_x(const float* __restrict__ x, _Float16* __restrict__ xf) {
  int idx = blockIdx.x * blockDim.x + threadIdx.x;
  if (idx >= M * DIMc) return;
  int c = idx & (DIMc - 1);
  int t = idx >> 7;
  int s = t & 7, p = t >> 3;
  int b = p / (Hh * Wd);
  int rem = p - b * (Hh * Wd);
  int hh = rem / Wd, ww = rem - hh * Wd;
  size_t xi = ((((size_t)b * NFr + s) * DIMc + c) * Hh + hh) * Wd + ww;
  xf[idx] = (_Float16)x[xi];
}

// layernorm over HID=256 features; optional sequence flip on the read side
__global__ __launch_bounds__(256) void k_layernorm(const float* __restrict__ h,
                                                   const float* __restrict__ nw,
                                                   const float* __restrict__ nb,
                                                   _Float16* __restrict__ outp,
                                                   int flip) {
  __shared__ float sh[256];
  int t_out = blockIdx.x;
  int p = t_out >> 3, s = t_out & 7;
  int t_in = (p << 3) | (flip ? (7 - s) : s);
  int c = threadIdx.x;
  float v = h[(size_t)t_in * HID + c];
  sh[c] = v; __syncthreads();
  for (int o = 128; o > 0; o >>= 1) { if (c < o) sh[c] += sh[c + o]; __syncthreads(); }
  float mu = sh[0] * (1.f / HID);
  __syncthreads();
  float d = v - mu;
  sh[c] = d * d; __syncthreads();
  for (int o = 128; o > 0; o >>= 1) { if (c < o) sh[c] += sh[c + o]; __syncthreads(); }
  float var = sh[0] * (1.f / HID);
  outp[(size_t)t_out * HID + c] = (_Float16)(d * rsqrtf(var + 1e-5f) * nw[c] + nb[c]);
}

// ---------------------------------------------------------------------------
// WMMA GEMM kernels (one wave = 32x64 macro tile, 8 waves/block)
// ---------------------------------------------------------------------------
__global__ __launch_bounds__(256) void k_gemm_gelu(const _Float16* __restrict__ A,
                                                   const _Float16* __restrict__ W,
                                                   const float* __restrict__ bias,
                                                   _Float16* __restrict__ outp,
                                                   int N, int K) {
  int wid = (blockIdx.x * 256 + threadIdx.x) >> 5;
  int ntb = N >> 6;
  int tm = wid / ntb, tn = wid - tm * ntb;
  if (tm >= M / (16 * MT)) return;
  v8f acc[MT][NT];
  gemm_block(A, W, tm * 16 * MT, tn * 16 * NT, K, acc);
  int lane = threadIdx.x & 31;
  int ro = (lane >> 4) * 8, co = lane & 15;
#pragma unroll
  for (int j = 0; j < NT; j++) {
    int col = tn * 16 * NT + j * 16 + co;
    float bb = bias[col];
#pragma unroll
    for (int i = 0; i < MT; i++) {
#pragma unroll
      for (int v = 0; v < 8; v++) {
        int row = tm * 16 * MT + i * 16 + ro + v;
        float val = acc[i][j][v] + bb;
        outp[(size_t)row * N + col] = (_Float16)geluf(val);
      }
    }
  }
}

__global__ __launch_bounds__(256) void k_gemm_f32(const _Float16* __restrict__ A,
                                                  const _Float16* __restrict__ W,
                                                  const float* __restrict__ bias,
                                                  float* __restrict__ outp,
                                                  int N, int K) {
  int wid = (blockIdx.x * 256 + threadIdx.x) >> 5;
  int ntb = N >> 6;
  int tm = wid / ntb, tn = wid - tm * ntb;
  if (tm >= M / (16 * MT)) return;
  v8f acc[MT][NT];
  gemm_block(A, W, tm * 16 * MT, tn * 16 * NT, K, acc);
  int lane = threadIdx.x & 31;
  int ro = (lane >> 4) * 8, co = lane & 15;
#pragma unroll
  for (int j = 0; j < NT; j++) {
    int col = tn * 16 * NT + j * 16 + co;
    float bb = bias[col];
#pragma unroll
    for (int i = 0; i < MT; i++) {
#pragma unroll
      for (int v = 0; v < 8; v++) {
        int row = tm * 16 * MT + i * 16 + ro + v;
        outp[(size_t)row * N + col] = acc[i][j][v] + bb;
      }
    }
  }
}

// up-projection: N=1024; cols [0,512) = xm -> gate_in section (stride 1536, +1024),
// cols [512,1024) = z buffer
__global__ __launch_bounds__(256) void k_gemm_up(const _Float16* __restrict__ A,
                                                 const _Float16* __restrict__ W,
                                                 const float* __restrict__ bias,
                                                 _Float16* __restrict__ G,
                                                 _Float16* __restrict__ Z) {
  constexpr int N = 2 * INNER, K = HID;
  int wid = (blockIdx.x * 256 + threadIdx.x) >> 5;
  int ntb = N >> 6;
  int tm = wid / ntb, tn = wid - tm * ntb;
  if (tm >= M / (16 * MT)) return;
  v8f acc[MT][NT];
  gemm_block(A, W, tm * 16 * MT, tn * 16 * NT, K, acc);
  int lane = threadIdx.x & 31;
  int ro = (lane >> 4) * 8, co = lane & 15;
#pragma unroll
  for (int j = 0; j < NT; j++) {
    int col = tn * 16 * NT + j * 16 + co;
    float bb = bias[col];
#pragma unroll
    for (int i = 0; i < MT; i++) {
#pragma unroll
      for (int v = 0; v < 8; v++) {
        int row = tm * 16 * MT + i * 16 + ro + v;
        float val = acc[i][j][v] + bb;
        if (col < INNER)
          G[(size_t)row * (3 * INNER) + 2 * INNER + col] = (_Float16)val;  // xm (= v matrix)
        else
          Z[(size_t)row * INNER + (col - INNER)] = (_Float16)val;          // z
      }
    }
  }
}

// down-projection: N=256, K=512; accumulate into residual h with sequence flip
__global__ __launch_bounds__(256) void k_gemm_down(const _Float16* __restrict__ A,
                                                   const _Float16* __restrict__ W,
                                                   const float* __restrict__ bias,
                                                   float* __restrict__ h, int flip) {
  constexpr int N = HID, K = INNER;
  int wid = (blockIdx.x * 256 + threadIdx.x) >> 5;
  int ntb = N >> 6;
  int tm = wid / ntb, tn = wid - tm * ntb;
  if (tm >= M / (16 * MT)) return;
  v8f acc[MT][NT];
  gemm_block(A, W, tm * 16 * MT, tn * 16 * NT, K, acc);
  int lane = threadIdx.x & 31;
  int ro = (lane >> 4) * 8, co = lane & 15;
#pragma unroll
  for (int j = 0; j < NT; j++) {
    int col = tn * 16 * NT + j * 16 + co;
    float bb = bias[col];
#pragma unroll
    for (int i = 0; i < MT; i++) {
#pragma unroll
      for (int v = 0; v < 8; v++) {
        int row = tm * 16 * MT + i * 16 + ro + v;
        int p = row >> 3, s = row & 7;
        int s2 = flip ? (7 - s) : s;
        h[(size_t)((p << 3) | s2) * HID + col] += acc[i][j][v] + bb;
      }
    }
  }
}

// final: N=128, K=256; out = x + o, scattered back to (B,NF,C,H,W)
__global__ __launch_bounds__(256) void k_gemm_final(const _Float16* __restrict__ A,
                                                    const _Float16* __restrict__ W,
                                                    const float* __restrict__ bias,
                                                    const float* __restrict__ x,
                                                    float* __restrict__ outp) {
  constexpr int N = DIMc, K = HID;
  int wid = (blockIdx.x * 256 + threadIdx.x) >> 5;
  int ntb = N >> 6;
  int tm = wid / ntb, tn = wid - tm * ntb;
  if (tm >= M / (16 * MT)) return;
  v8f acc[MT][NT];
  gemm_block(A, W, tm * 16 * MT, tn * 16 * NT, K, acc);
  int lane = threadIdx.x & 31;
  int ro = (lane >> 4) * 8, co = lane & 15;
#pragma unroll
  for (int j = 0; j < NT; j++) {
    int col = tn * 16 * NT + j * 16 + co;
    float bb = bias[col];
#pragma unroll
    for (int i = 0; i < MT; i++) {
#pragma unroll
      for (int v = 0; v < 8; v++) {
        int row = tm * 16 * MT + i * 16 + ro + v;
        int s = row & 7, p = row >> 3;
        int b = p / (Hh * Wd);
        int rem = p - b * (Hh * Wd);
        int hh = rem / Wd, ww = rem - hh * Wd;
        size_t xi = ((((size_t)b * NFr + s) * DIMc + col) * Hh + hh) * Wd + ww;
        outp[xi] = x[xi] + acc[i][j][v] + bb;
      }
    }
  }
}

// ---------------------------------------------------------------------------
// Per-(pixel,head) VALU kernels: shapes (DH=4, S=8) are far below WMMA tiles
// ---------------------------------------------------------------------------
// causal depthwise conv(k=4) + silu + per-head 4x4 q/k projections
__global__ __launch_bounds__(256) void k_conv_qk(_Float16* __restrict__ G,
                                                 _Float16* __restrict__ XA,
                                                 const float* __restrict__ convw,
                                                 const float* __restrict__ convb,
                                                 const float* __restrict__ qw,
                                                 const float* __restrict__ kw) {
  int idx = blockIdx.x * blockDim.x + threadIdx.x;
  if (idx >= NPIX * NH) return;
  int hd = idx & (NH - 1), p = idx >> 7;

  float xm[8][4], xa[8][4];
#pragma unroll
  for (int s = 0; s < 8; s++) {
    size_t t = (size_t)((p << 3) | s);
#pragma unroll
    for (int d = 0; d < 4; d++)
      xm[s][d] = (float)G[t * (3 * INNER) + 2 * INNER + hd * 4 + d];
  }
#pragma unroll
  for (int d = 0; d < 4; d++) {
    int ch = hd * 4 + d;
    float w0 = convw[ch * 4 + 0], w1 = convw[ch * 4 + 1];
    float w2 = convw[ch * 4 + 2], w3 = convw[ch * 4 + 3];
    float cb = convb[ch];
#pragma unroll
    for (int s = 0; s < 8; s++) {
      float acc = cb + w3 * xm[s][d];
      if (s >= 1) acc += w2 * xm[s - 1][d];
      if (s >= 2) acc += w1 * xm[s - 2][d];
      if (s >= 3) acc += w0 * xm[s - 3][d];
      float sv = siluf(acc);
      xa[s][d] = sv;
      XA[(size_t)((p << 3) | s) * INNER + ch] = (_Float16)sv;
    }
  }
  // q/k: q[e] = sum_d xa[d] * qw[hd][e][d]
#pragma unroll
  for (int s = 0; s < 8; s++) {
    size_t t = (size_t)((p << 3) | s);
#pragma unroll
    for (int e = 0; e < 4; e++) {
      float qa = 0.f, ka = 0.f;
#pragma unroll
      for (int d = 0; d < 4; d++) {
        qa += xa[s][d] * qw[(hd * 4 + e) * 4 + d];
        ka += xa[s][d] * kw[(hd * 4 + e) * 4 + d];
      }
      G[t * (3 * INNER) + hd * 4 + e]         = (_Float16)qa;
      G[t * (3 * INNER) + INNER + hd * 4 + e] = (_Float16)ka;
    }
  }
}

// mLSTM (S=8, DH=4) + head-norm + skip + silu(z) gate -> f16 input for down-GEMM
__global__ __launch_bounds__(256) void k_mlstm(const _Float16* __restrict__ G,
                                               const float* __restrict__ igfg,
                                               const _Float16* __restrict__ XA,
                                               const _Float16* __restrict__ Z,
                                               const float* __restrict__ onw,
                                               const float* __restrict__ onb,
                                               const float* __restrict__ skp,
                                               _Float16* __restrict__ HS) {
  int idx = blockIdx.x * blockDim.x + threadIdx.x;
  if (idx >= NPIX * NH) return;
  int hd = idx & (NH - 1), p = idx >> 7;

  float q[8][4], k[8][4], v[8][4], ig[8], lfc[8];
  float lf = 0.f;
#pragma unroll
  for (int s = 0; s < 8; s++) {
    size_t t = (size_t)((p << 3) | s);
#pragma unroll
    for (int d = 0; d < 4; d++) {
      q[s][d] = (float)G[t * (3 * INNER) + hd * 4 + d];
      k[s][d] = (float)G[t * (3 * INNER) + INNER + hd * 4 + d];
      v[s][d] = (float)G[t * (3 * INNER) + 2 * INNER + hd * 4 + d];
    }
    ig[s] = igfg[t * 256 + hd];
    float fg = igfg[t * 256 + 128 + hd];
    float ls = fminf(fg, 0.f) - log1pf(expf(-fabsf(fg)));  // log_sigmoid
    lf += ls;
    lfc[s] = lf;
  }
#pragma unroll
  for (int s = 0; s < 8; s++) {
    float logD[8], maxD = -3.402823466e38f;
#pragma unroll
    for (int tt = 0; tt < 8; tt++) {
      if (tt <= s) {
        logD[tt] = lfc[s] - lfc[tt] + ig[tt];
        maxD = fmaxf(maxD, logD[tt]);
      }
    }
    float c[8], csum = 0.f;
#pragma unroll
    for (int tt = 0; tt < 8; tt++) {
      if (tt <= s) {
        float qk = 0.f;
#pragma unroll
        for (int d = 0; d < 4; d++) qk += q[s][d] * k[tt][d];
        qk *= 0.5f;  // 1/sqrt(DH=4)
        float val = qk * expf(logD[tt] - maxD);
        c[tt] = val;
        csum += val;
      }
    }
    float norm = fmaxf(fabsf(csum), expf(-maxD)) + 1e-6f;
    float hs4[4] = {0.f, 0.f, 0.f, 0.f};
#pragma unroll
    for (int tt = 0; tt < 8; tt++) {
      if (tt <= s) {
        float w = c[tt] / norm;
#pragma unroll
        for (int d = 0; d < 4; d++) hs4[d] += w * v[tt][d];
      }
    }
    // head layernorm over DH=4
    float mu = (hs4[0] + hs4[1] + hs4[2] + hs4[3]) * 0.25f;
    float var = 0.f;
#pragma unroll
    for (int d = 0; d < 4; d++) { float dd = hs4[d] - mu; var += dd * dd; }
    var *= 0.25f;
    float rstd = rsqrtf(var + 1e-5f);
    size_t t = (size_t)((p << 3) | s);
#pragma unroll
    for (int d = 0; d < 4; d++) {
      int ch = hd * 4 + d;
      float hn = (hs4[d] - mu) * rstd * onw[ch] + onb[ch];
      float hv = hn + skp[ch] * (float)XA[t * INNER + ch];
      float z = (float)Z[t * INNER + ch];
      HS[t * INNER + ch] = (_Float16)(hv * siluf(z));
    }
  }
}

}  // namespace vil

// ---------------------------------------------------------------------------
extern "C" void kernel_launch(void* const* d_in, const int* in_sizes, int n_in,
                              void* d_out, int out_size, void* d_ws, size_t ws_size,
                              hipStream_t stream) {
  using namespace vil;
  const float* x       = (const float*)d_in[0];
  const float* fc1_w   = (const float*)d_in[1];
  const float* fc1_b   = (const float*)d_in[2];
  const float* fc2_w   = (const float*)d_in[3];
  const float* fc2_b   = (const float*)d_in[4];
  const float* ofc1_w  = (const float*)d_in[5];
  const float* ofc1_b  = (const float*)d_in[6];
  const float* ofc2_w  = (const float*)d_in[7];
  const float* ofc2_b  = (const float*)d_in[8];
  const float* norm_w  = (const float*)d_in[9];
  const float* norm_b  = (const float*)d_in[10];
  const float* up_w    = (const float*)d_in[11];
  const float* up_b    = (const float*)d_in[12];
  const float* conv_w  = (const float*)d_in[13];
  const float* conv_b  = (const float*)d_in[14];
  const float* q_w     = (const float*)d_in[15];
  const float* k_w     = (const float*)d_in[16];
  const float* ig_w    = (const float*)d_in[17];
  const float* ig_b    = (const float*)d_in[18];
  const float* fg_w    = (const float*)d_in[19];
  const float* fg_b    = (const float*)d_in[20];
  const float* onorm_w = (const float*)d_in[21];
  const float* onorm_b = (const float*)d_in[22];
  const float* skp     = (const float*)d_in[23];
  const float* down_w  = (const float*)d_in[24];
  const float* down_b  = (const float*)d_in[25];
  float* outp = (float*)d_out;

  char* ws = (char*)d_ws;
  size_t off = 0;
  auto alloc = [&](size_t bytes) -> void* {
    void* p = ws + off;
    off = (off + bytes + 255) & ~(size_t)255;
    return p;
  };

  // f16 weight copies (N,K row-major == reference layout)
  _Float16* Wfc1    = (_Float16*)alloc((size_t)DIMc * DIMc * 2);
  _Float16* Wfc2    = (_Float16*)alloc((size_t)HID * DIMc * 2);
  _Float16* Wout1   = (_Float16*)alloc((size_t)HID * HID * 2);
  _Float16* Wout2   = (_Float16*)alloc((size_t)DIMc * HID * 2);
  _Float16* Wup[2]  = {(_Float16*)alloc((size_t)2 * INNER * HID * 2),
                       (_Float16*)alloc((size_t)2 * INNER * HID * 2)};
  _Float16* Wgate[2] = {(_Float16*)alloc((size_t)256 * 1536 * 2),
                        (_Float16*)alloc((size_t)256 * 1536 * 2)};
  _Float16* Wdown[2] = {(_Float16*)alloc((size_t)HID * INNER * 2),
                        (_Float16*)alloc((size_t)HID * INNER * 2)};
  float* gateBias = (float*)alloc(256 * 4);

  // activations
  _Float16* xflat = (_Float16*)alloc((size_t)M * DIMc * 2);
  _Float16* D1    = (_Float16*)alloc((size_t)M * DIMc * 2);
  float*    h     = (float*)alloc((size_t)M * HID * 4);
  _Float16* ynorm = (_Float16*)alloc((size_t)M * HID * 2);   // also f16(h) before out_fc1
  _Float16* C2    = (_Float16*)alloc((size_t)M * HID * 2);
  _Float16* G     = (_Float16*)alloc((size_t)M * 3 * INNER * 2);  // [q | k | xm] gate_in
  _Float16* Zb    = (_Float16*)alloc((size_t)M * INNER * 2);
  _Float16* XA    = (_Float16*)alloc((size_t)M * INNER * 2);
  float*    igfg  = (float*)alloc((size_t)M * 256 * 4);
  _Float16* HS    = (_Float16*)alloc((size_t)M * INNER * 2);

  auto cvt = [&](const float* s, _Float16* d, int n) {
    k_cvt16<<<(n + 255) / 256, 256, 0, stream>>>(s, d, n);
  };
  // waves = (M / 32) * (N / 64); 8 waves per block
  auto gemm_blocks = [&](int N) { return ((M / (16 * MT)) * (N / (16 * NT))) / 8; };

  // weight conversions
  cvt(fc1_w, Wfc1, DIMc * DIMc);
  cvt(fc2_w, Wfc2, HID * DIMc);
  cvt(ofc1_w, Wout1, HID * HID);
  cvt(ofc2_w, Wout2, DIMc * HID);
  for (int i = 0; i < 2; i++) {
    cvt(up_w + (size_t)i * 2 * INNER * HID, Wup[i], 2 * INNER * HID);
    cvt(ig_w + (size_t)i * NH * 1536, Wgate[i], NH * 1536);
    cvt(fg_w + (size_t)i * NH * 1536, Wgate[i] + (size_t)NH * 1536, NH * 1536);
    cvt(down_w + (size_t)i * HID * INNER, Wdown[i], HID * INNER);
  }

  // x -> x_flat (f16)
  k_pack_x<<<(M * DIMc + 255) / 256, 256, 0, stream>>>(x, xflat);

  // in_fc1 (gelu) and in_fc2 -> h
  k_gemm_gelu<<<gemm_blocks(DIMc), 256, 0, stream>>>(xflat, Wfc1, fc1_b, D1, DIMc, DIMc);
  k_gemm_f32<<<gemm_blocks(HID), 256, 0, stream>>>(D1, Wfc2, fc2_b, h, HID, DIMc);

  const int blkPH = (NPIX * NH + 255) / 256;
  for (int i = 0; i < 2; i++) {
    const int flip = i;  // layer 1 runs on reversed sequence
    k_layernorm<<<M, 256, 0, stream>>>(h, norm_w + i * HID, norm_b + i * HID, ynorm, flip);
    k_gemm_up<<<gemm_blocks(2 * INNER), 256, 0, stream>>>(ynorm, Wup[i],
                                                          up_b + i * 2 * INNER, G, Zb);
    k_conv_qk<<<blkPH, 256, 0, stream>>>(G, XA, conv_w + i * INNER * 4,
                                         conv_b + i * INNER,
                                         q_w + i * NH * 16, k_w + i * NH * 16);
    // combined ig||fg bias, then gates GEMM (K=1536, N=256)
    k_copy_f32<<<1, 128, 0, stream>>>(ig_b + i * NH, gateBias, NH);
    k_copy_f32<<<1, 128, 0, stream>>>(fg_b + i * NH, gateBias + NH, NH);
    k_gemm_f32<<<gemm_blocks(256), 256, 0, stream>>>(G, Wgate[i], gateBias, igfg, 256, 1536);
    k_mlstm<<<blkPH, 256, 0, stream>>>(G, igfg, XA, Zb,
                                       onorm_w + i * INNER, onorm_b + i * INNER,
                                       skp + i * INNER, HS);
    k_gemm_down<<<gemm_blocks(HID), 256, 0, stream>>>(HS, Wdown[i], down_b + i * HID, h, flip);
  }

  // output MLP + residual scatter back to (B,NF,C,H,W)
  cvt(h, ynorm, M * HID);
  k_gemm_gelu<<<gemm_blocks(HID), 256, 0, stream>>>(ynorm, Wout1, ofc1_b, C2, HID, HID);
  k_gemm_final<<<gemm_blocks(DIMc), 256, 0, stream>>>(C2, Wout2, ofc2_b, x, outp);
}
